// GCN_88725434400874
// MI455X (gfx1250) — matrix operand order repaired and checked by reading, not compiled, not run on hardware
//
#include <hip/hip_runtime.h>

// MI455X / gfx1250 GCN forward.
//  - Node GEMMs: one wave per 16-node tile, single v_wmma_f32_16x16x32_f16
//    (two for Fout=32), K zero-padded via pre-packed f16 B operands in the
//    exact WMMA B-lane layout -> unconditional vector loads, no divergence.
//  - Aggregation: global_atomic_add_f32 scatter (node tables are L2-resident).
//  - Edge MLP collapsed: concat@lw1@lw2 == h[src]@Wc_top + h[dst]@Wc_bot + bc,
//    computed per-node (100K rows) instead of per-edge (1.6M rows).

typedef __attribute__((ext_vector_type(16))) _Float16 v16h;
typedef __attribute__((ext_vector_type(8)))  float    v8f;
typedef _Float16 half_t;

#define NPAD 100352  // padded node rows in workspace (>= ceil(N/16)*16)

__global__ void fill_f32(float* __restrict__ p, float v, int n) {
    int i = blockIdx.x * blockDim.x + threadIdx.x;
    if (i < n) p[i] = v;
}

// deg starts at 1.0 (self loop); each undirected edge hits both endpoints
// == in-degree over the concatenated bidirectional edge_index.
__global__ void deg_count(const int* __restrict__ e0, const int* __restrict__ e1,
                          float* __restrict__ deg, int E) {
    int e = blockIdx.x * blockDim.x + threadIdx.x;
    if (e >= E) return;
    atomicAdd(&deg[e0[e]], 1.0f);
    atomicAdd(&deg[e1[e]], 1.0f);
}

__global__ void make_dinv(float* __restrict__ d, int N) {
    int i = blockIdx.x * blockDim.x + threadIdx.x;
    if (i < N) d[i] = rsqrtf(d[i]);
}

// Wc[64x3] = lw1[64x4] @ lw2[4x3]; bc[3] = lb1 @ lw2 + lb2 (no ReLU between).
__global__ void combine_w(const float* __restrict__ lw1, const float* __restrict__ lb1,
                          const float* __restrict__ lw2, const float* __restrict__ lb2,
                          float* __restrict__ Wc, float* __restrict__ bc) {
    int t = threadIdx.x;
    if (t < 192) {
        int r = t / 3, c = t % 3;
        float s = 0.f;
        #pragma unroll
        for (int j = 0; j < 4; ++j) s += lw1[r * 4 + j] * lw2[j * 3 + c];
        Wc[t] = s;
    } else if (t < 195) {
        int c = t - 192;
        float s = lb2[c];
        #pragma unroll
        for (int j = 0; j < 4; ++j) s += lb1[j] * lw2[j * 3 + c];
        bc[c] = s;
    }
}

// Stage x [N,7] into zero-padded stride-8 buffer (keeps GEMM A loads in-bounds).
__global__ void pad_x(const float* __restrict__ x, float* __restrict__ xp, int N) {
    int i = blockIdx.x * blockDim.x + threadIdx.x;
    if (i >= N * 8) return;
    int n = i >> 3, k = i & 7;
    xp[i] = (k < 7) ? x[n * 7 + k] : 0.f;
}

// Pre-pack all B operands as f16 in the WMMA B-lane layout (B[32x16]:
// N = lane&15, K = e<8 ? (lane>>4)*8 + e : 16 + (lane>>4)*8 + (e-8)), with compile-time zero K/N padding.
//   mat0: W1 (7x8)        mat1: W2 (8x16)
//   mat2: W3 cols 0-15    mat3: W3 cols 16-31   (Fin=16)
//   mat4: combined MLP Wc: cols 0-2 = Wc[0:32], cols 4-6 = Wc[32:64]
__global__ void pack_B(const float* __restrict__ W1, const float* __restrict__ W2,
                       const float* __restrict__ W3, const float* __restrict__ Wc,
                       half_t* __restrict__ Bpk) {
    int idx = blockIdx.x * blockDim.x + threadIdx.x;
    if (idx >= 5 * 512) return;
    int mat = idx >> 9, t = idx & 511;
    int lane = t >> 4, e = t & 15;
    int m = lane & 15, hi = lane >> 4;
    int k = (e < 8) ? (hi * 8 + e) : (16 + hi * 8 + (e - 8));
    float v = 0.f;
    if (mat == 0)      { if (k < 7  && m < 8) v = W1[k * 8 + m]; }
    else if (mat == 1) { if (k < 8)           v = W2[k * 16 + m]; }
    else if (mat == 2) { if (k < 16)          v = W3[k * 32 + m]; }
    else if (mat == 3) { if (k < 16)          v = W3[k * 32 + 16 + m]; }
    else {
        if (m < 3)                 v = Wc[k * 3 + m];
        else if (m >= 4 && m < 7)  v = Wc[(32 + k) * 3 + (m - 4)];
    }
    Bpk[idx] = (half_t)v;
}

// xw[N,FOUT] = hin[N,FIN] @ W.  One wave per 16-node tile.
// A layout (16-bit 16x32, ISA 7.12.2): lane half hi: e0-7 -> K=hi*8+0..7,
// e8-15 -> K=hi*8+16..23 (both contiguous -> float4 loads). K >= FIN lanes
// load in-buffer garbage that is annihilated by zero B rows.
template <int FIN, int FOUT>
__global__ __launch_bounds__(256) void gemm_node(
    const float* __restrict__ hin, const half_t* __restrict__ B0,
    const half_t* __restrict__ B1, float* __restrict__ xw, int numTiles)
{
    const int lane = threadIdx.x & 31;
    const int tile = blockIdx.x * (blockDim.x >> 5) + (threadIdx.x >> 5);
    if (tile >= numTiles) return;           // wave-uniform: EXEC all-1s at WMMA
    const int m  = lane & 15;
    const int hi = lane >> 4;
    const int node = tile * 16 + m;

    const float4* r4 = reinterpret_cast<const float4*>(hin + (size_t)node * FIN);
    float4 c0 = r4[hi * 2 + 0];
    float4 c1 = r4[hi * 2 + 1];
    float4 c2 = r4[hi * 2 + 4];
    float4 c3 = r4[hi * 2 + 5];
    v16h a;
    a[0]  = (half_t)c0.x; a[1]  = (half_t)c0.y; a[2]  = (half_t)c0.z; a[3]  = (half_t)c0.w;
    a[4]  = (half_t)c1.x; a[5]  = (half_t)c1.y; a[6]  = (half_t)c1.z; a[7]  = (half_t)c1.w;
    a[8]  = (half_t)c2.x; a[9]  = (half_t)c2.y; a[10] = (half_t)c2.z; a[11] = (half_t)c2.w;
    a[12] = (half_t)c3.x; a[13] = (half_t)c3.y; a[14] = (half_t)c3.z; a[15] = (half_t)c3.w;

    v16h b0 = *reinterpret_cast<const v16h*>(B0 + lane * 16);
    v8f acc0 = {};
    acc0 = __builtin_amdgcn_wmma_f32_16x16x32_f16(false, a, false, b0,
                                                  (short)0, acc0, false, false);
    if constexpr (FOUT <= 16) {
        // D: VGPR r -> M = r + 8*hi, N = lane&15.
        #pragma unroll
        for (int r = 0; r < 8; ++r)
            xw[(size_t)(tile * 16 + hi * 8 + r) * FOUT + m] = acc0[r];
    } else {
        v16h b1 = *reinterpret_cast<const v16h*>(B1 + lane * 16);
        v8f acc1 = {};
        acc1 = __builtin_amdgcn_wmma_f32_16x16x32_f16(false, a, false, b1,
                                                      (short)0, acc1, false, false);
        #pragma unroll
        for (int r = 0; r < 8; ++r) {
            size_t row = (size_t)(tile * 16 + hi * 8 + r);
            xw[row * 32 + m]      = acc0[r];
            xw[row * 32 + 16 + m] = acc1[r];
        }
    }
}

// msg[dst] += xw[src]*dinv[src]*dinv[dst], both directions per undirected edge.
template <int F>
__global__ void scatter_edges(const int* __restrict__ e0, const int* __restrict__ e1,
                              const float* __restrict__ xw, const float* __restrict__ dinv,
                              float* __restrict__ msg, int E) {
    int e = blockIdx.x * blockDim.x + threadIdx.x;
    if (e >= E) return;
    int a = e0[e], b = e1[e];
    float nrm = dinv[a] * dinv[b];
    const float4* xa = reinterpret_cast<const float4*>(xw + (size_t)a * F);
    const float4* xb = reinterpret_cast<const float4*>(xw + (size_t)b * F);
    float* ma = msg + (size_t)a * F;
    float* mb = msg + (size_t)b * F;
    #pragma unroll
    for (int c4 = 0; c4 < (F >> 2); ++c4) {
        float4 va = xa[c4];
        float4 vb = xb[c4];
        int base = c4 * 4;
        atomicAdd(mb + base + 0, va.x * nrm);
        atomicAdd(mb + base + 1, va.y * nrm);
        atomicAdd(mb + base + 2, va.z * nrm);
        atomicAdd(mb + base + 3, va.w * nrm);
        atomicAdd(ma + base + 0, vb.x * nrm);
        atomicAdd(ma + base + 1, vb.y * nrm);
        atomicAdd(ma + base + 2, vb.z * nrm);
        atomicAdd(ma + base + 3, vb.w * nrm);
    }
}

// h = relu(msg + xw*dinv^2 + bias)
template <int F>
__global__ void finalize(const float* __restrict__ msg, const float* __restrict__ xw,
                         const float* __restrict__ dinv, const float* __restrict__ bias,
                         float* __restrict__ hout, int N) {
    int i = blockIdx.x * blockDim.x + threadIdx.x;
    if (i >= N * F) return;
    int n = i / F, f = i % F;               // F is a power of two -> shifts
    float di = dinv[n];
    float v = msg[i] + xw[i] * di * di + bias[f];
    hout[i] = fmaxf(v, 0.f);
}

// out[e, c] = p[src, c] + p[dst, 4+c] + bc[c]   (p stride 16)
__global__ void edge_out(const int* __restrict__ e0, const int* __restrict__ e1,
                         const float* __restrict__ p, const float* __restrict__ bc,
                         float* __restrict__ out, int E) {
    int e = blockIdx.x * blockDim.x + threadIdx.x;
    if (e >= E) return;
    int a = e0[e], b = e1[e];
    float4 ps = *reinterpret_cast<const float4*>(p + (size_t)a * 16);
    float4 pd = *reinterpret_cast<const float4*>(p + (size_t)b * 16 + 4);
    out[(size_t)e * 3 + 0] = ps.x + pd.x + bc[0];
    out[(size_t)e * 3 + 1] = ps.y + pd.y + bc[1];
    out[(size_t)e * 3 + 2] = ps.z + pd.z + bc[2];
}

static inline int cdiv(int a, int b) { return (a + b - 1) / b; }

extern "C" void kernel_launch(void* const* d_in, const int* in_sizes, int n_in,
                              void* d_out, int out_size, void* d_ws, size_t ws_size,
                              hipStream_t stream) {
    const float* x   = (const float*)d_in[0];
    // d_in[1] edge_index is cat(w2b, reversed) -> equivalent info in w2b.
    const int*   w2b = (const int*)d_in[2];
    const float* W1  = (const float*)d_in[3];
    const float* b1  = (const float*)d_in[4];
    const float* W2  = (const float*)d_in[5];
    const float* b2  = (const float*)d_in[6];
    const float* W3  = (const float*)d_in[7];
    const float* b3  = (const float*)d_in[8];
    const float* lw1 = (const float*)d_in[9];
    const float* lb1 = (const float*)d_in[10];
    const float* lw2 = (const float*)d_in[11];
    const float* lb2 = (const float*)d_in[12];

    const int N = in_sizes[0] / 7;      // 100000
    const int E = in_sizes[2] / 2;      // 1600000 undirected w2b edges
    const int* e0 = w2b;
    const int* e1 = w2b + E;

    // Workspace layout (floats): ~46 MB.
    float*  ws    = (float*)d_ws;
    float*  dinv  = ws;                              // N (deg -> rsqrt in place)
    float*  bufIn = ws    + (size_t)NPAD;            // x-pad / h (rotates in place)
    float*  bufXw = bufIn + (size_t)NPAD * 32;       // xw
    float*  bufMsg= bufXw + (size_t)NPAD * 32;       // msg accumulator
    float*  p     = bufMsg+ (size_t)NPAD * 32;       // per-node projections [N x 16]
    float*  Wc    = p     + (size_t)NPAD * 16;       // 192
    float*  bc    = Wc + 192;                        // 3 (+pad to 64)
    half_t* Bpk   = (half_t*)(bc + 64);              // 5 x 512 f16, 32B-aligned
    float*  out   = (float*)d_out;

    const int tiles = cdiv(N, 16);
    const int gblk  = cdiv(tiles, 8);                // 8 waves (tiles) per block

    // Degree / normalization + weight packing.
    fill_f32 <<<cdiv(N, 256), 256, 0, stream>>>(dinv, 1.0f, N);
    deg_count<<<cdiv(E, 256), 256, 0, stream>>>(e0, e1, dinv, E);
    make_dinv<<<cdiv(N, 256), 256, 0, stream>>>(dinv, N);
    combine_w<<<1, 256, 0, stream>>>(lw1, lb1, lw2, lb2, Wc, bc);
    pack_B   <<<10, 256, 0, stream>>>(W1, W2, W3, Wc, Bpk);
    pad_x    <<<cdiv(N * 8, 256), 256, 0, stream>>>(x, bufIn, N);

    // Layer 1: 7(->8 padded) -> 8
    fill_f32        <<<cdiv(N * 8, 256), 256, 0, stream>>>(bufMsg, 0.f, N * 8);
    gemm_node<8, 8> <<<gblk, 256, 0, stream>>>(bufIn, Bpk + 0 * 512, Bpk, bufXw, tiles);
    scatter_edges<8><<<cdiv(E, 256), 256, 0, stream>>>(e0, e1, bufXw, dinv, bufMsg, E);
    finalize<8>     <<<cdiv(N * 8, 256), 256, 0, stream>>>(bufMsg, bufXw, dinv, b1, bufIn, N);

    // Layer 2: 8 -> 16
    fill_f32         <<<cdiv(N * 16, 256), 256, 0, stream>>>(bufMsg, 0.f, N * 16);
    gemm_node<8, 16> <<<gblk, 256, 0, stream>>>(bufIn, Bpk + 1 * 512, Bpk, bufXw, tiles);
    scatter_edges<16><<<cdiv(E, 256), 256, 0, stream>>>(e0, e1, bufXw, dinv, bufMsg, E);
    finalize<16>     <<<cdiv(N * 16, 256), 256, 0, stream>>>(bufMsg, bufXw, dinv, b2, bufIn, N);

    // Layer 3: 16 -> 32 (two WMMAs per tile: cols 0-15 and 16-31)
    fill_f32         <<<cdiv(N * 32, 256), 256, 0, stream>>>(bufMsg, 0.f, N * 32);
    gemm_node<16, 32><<<gblk, 256, 0, stream>>>(bufIn, Bpk + 2 * 512, Bpk + 3 * 512, bufXw, tiles);
    scatter_edges<32><<<cdiv(E, 256), 256, 0, stream>>>(e0, e1, bufXw, dinv, bufMsg, E);
    finalize<32>     <<<cdiv(N * 32, 256), 256, 0, stream>>>(bufMsg, bufXw, dinv, b3, bufIn, N);

    // Collapsed edge-MLP: per-node src/dst projections (one WMMA), then edges.
    gemm_node<32, 16><<<gblk, 256, 0, stream>>>(bufIn, Bpk + 4 * 512, Bpk, p, tiles);
    edge_out<<<cdiv(E, 256), 256, 0, stream>>>(e0, e1, p, bc, out, E);
}
